// HierarchicalGraphBuilder_57775900066446
// MI455X (gfx1250) — compile-verified, matrix operand order checked by reading, not run on hardware
//
#include <hip/hip_runtime.h>
#include <hip/hip_bf16.h>
#include <math.h>

#define N_PTS 6144
#define FEAT  512
#define HID   256
#define KMAX  50
#define KNN   20
#define KPAD  64
#define NPART 8
#define COLS_PER_PART (N_PTS / NPART)        // 768
#define TILES_PER_PART (COLS_PER_PART / 16)  // 48
#define NGROUPS (TILES_PER_PART / 4)         // 12 groups of 4 tiles (64 cols)
#define THR   0.1f
#define SPATIAL_DECAY 50.0f

typedef __attribute__((ext_vector_type(16))) __bf16 v16bf;
typedef __attribute__((ext_vector_type(8)))  float  v8f;

// ---------- bf16 helpers ----------
__device__ inline unsigned short f2bf(float f) {
  unsigned int u = __float_as_uint(f);
  unsigned int rounding = 0x7FFFu + ((u >> 16) & 1u);
  return (unsigned short)((u + rounding) >> 16);
}
__device__ inline __bf16 bits2bf(unsigned short s) {
  union { unsigned short u; __bf16 b; } x; x.u = s; return x.b;
}

__device__ inline v8f wmma_bf16(v16bf a, v16bf b, v8f c) {
  // v_wmma_f32_16x16x32_bf16
  return __builtin_amdgcn_wmma_f32_16x16x32_bf16(
      /*neg_a=*/false, a, /*neg_b=*/false, b,
      /*c_mod=*/(short)0, c, /*reuse_a=*/false, /*reuse_b=*/false);
}

// A fragment (16x32 bf16, M x K): lane&15 = row, lane>>4 selects K half.
// element e: K = kbase + (e<8 ? e : e+8), kbase = 8*(lane>>4)
__device__ inline v16bf load_frag_row(const unsigned short* base, int ld,
                                      int row0, int k0, int lane) {
  int m  = row0 + (lane & 15);
  int kb = k0 + ((lane >> 4) << 3);
  const unsigned short* p = base + (size_t)m * ld + kb;
  v16bf r;
#pragma unroll
  for (int e = 0; e < 8; ++e) { r[e] = bits2bf(p[e]); r[e + 8] = bits2bf(p[e + 16]); }
  return r;
}

// B fragment (32x16 bf16, K x N, row-major B): lane&15 = column n.
__device__ inline v16bf load_frag_col(const unsigned short* base, int ld,
                                      int k0, int n0, int lane) {
  int n  = n0 + (lane & 15);
  int kb = k0 + ((lane >> 4) << 3);
  const unsigned short* p = base + (size_t)kb * ld + n;
  v16bf r;
#pragma unroll
  for (int e = 0; e < 8; ++e) {
    r[e]     = bits2bf(p[(size_t)e * ld]);
    r[e + 8] = bits2bf(p[(size_t)(e + 16) * ld]);
  }
  return r;
}

// ---------- utility kernels ----------
__global__ void fill_kernel(float* p, float v, int n) {
  int i = blockIdx.x * blockDim.x + threadIdx.x;
  if (i < n) p[i] = v;
}

__global__ void convW1_kernel(const float* W1, unsigned short* W1b) {
  int i = blockIdx.x * blockDim.x + threadIdx.x;
  if (i < FEAT * HID) W1b[i] = f2bf(W1[i]);
}

__global__ void convW2_kernel(const float* W2, unsigned short* W2b) {
  int i = blockIdx.x * blockDim.x + threadIdx.x;
  if (i < HID * KPAD) {
    int k = i / KPAD, n = i % KPAD;
    W2b[i] = (n < KMAX) ? f2bf(W2[k * KMAX + n]) : (unsigned short)0;
  }
}

// row L2-normalize -> fnb (bf16), raw copy -> Xb (bf16)
__global__ void prep_kernel(const float* feat, unsigned short* fnb, unsigned short* Xb) {
  __shared__ float red[256];
  int i = blockIdx.x, t = threadIdx.x;
  float ss = 0.f;
  for (int c = t; c < FEAT; c += 256) { float x = feat[(size_t)i * FEAT + c]; ss += x * x; }
  red[t] = ss; __syncthreads();
  for (int s = 128; s > 0; s >>= 1) { if (t < s) red[t] += red[t + s]; __syncthreads(); }
  float inv = 1.f / fmaxf(sqrtf(red[0]), 1e-12f);
  for (int c = t; c < FEAT; c += 256) {
    float x = feat[(size_t)i * FEAT + c];
    fnb[(size_t)i * FEAT + c] = f2bf(x * inv);
    Xb[(size_t)i * FEAT + c]  = f2bf(x);
  }
}

// ---------- fused sim GEMM + spatial kernel + sigmoid + streaming top-20 ----------
// Each wave: 16-row strip x 64-column group per pass. One A fragment feeds 4 B
// fragments / 4 independent accumulator chains (A traffic /4, WMMA issue overlap).
__global__ __launch_bounds__(32) void simtopk_kernel(
    const unsigned short* fnb, const float* pos,
    const float* lambda_weight, const float* temperature,
    float* cand_val, int* cand_idx) {
  __shared__ float tile[16][65];      // 64 cols + pad
  __shared__ float lv[16][KNN];
  __shared__ int   li[16][KNN];
  int lane  = threadIdx.x;
  int strip = blockIdx.x / NPART;
  int part  = blockIdx.x % NPART;
  int i0 = strip * 16;
  float lam   = 1.f / (1.f + expf(-lambda_weight[0]));
  float one_m = 1.f - lam;
  float temp  = temperature[0];
  const float inv2s2 = 1.0f / (2.0f * SPATIAL_DECAY * SPATIAL_DECAY);

  if (lane < 16)
    for (int q = 0; q < KNN; ++q) { lv[lane][q] = -1e30f; li[lane][q] = 0; }
  __syncthreads();

  int hi = lane >> 4, col = lane & 15;
  const v8f vzero = {0.f, 0.f, 0.f, 0.f, 0.f, 0.f, 0.f, 0.f};

  for (int g = 0; g < NGROUPS; ++g) {
    int j0 = part * COLS_PER_PART + g * 64;

    // warm L2 for next group's B rows (global_prefetch_b8); 2 rows per lane
    if (g + 1 < NGROUPS) {
      const unsigned short* pf = fnb + (size_t)(j0 + 64 + lane * 2) * FEAT;
      __builtin_prefetch(pf, 0, 1);
      __builtin_prefetch(pf + FEAT, 0, 1);
    }

    v8f acc[4];
#pragma unroll
    for (int t = 0; t < 4; ++t) acc[t] = vzero;

    for (int kt = 0; kt < FEAT; kt += 32) {
      v16bf a = load_frag_row(fnb, FEAT, i0, kt, lane);   // shared A fragment
#pragma unroll
      for (int t = 0; t < 4; ++t) {
        v16bf b = load_frag_row(fnb, FEAT, j0 + t * 16, kt, lane);  // B = fn.T
        acc[t] = wmma_bf16(a, b, acc[t]);
      }
    }

#pragma unroll
    for (int t = 0; t < 4; ++t) {
      int j = j0 + t * 16 + col;
      float pjx = pos[2 * j], pjy = pos[2 * j + 1];
#pragma unroll
      for (int r = 0; r < 8; ++r) {
        int i = i0 + r + 8 * hi;
        float dx = pos[2 * i] - pjx, dy = pos[2 * i + 1] - pjy;
        float d2 = dx * dx + dy * dy;
        float sk = expf(-d2 * inv2s2);
        float x = (lam * acc[t][r] + one_m * sk) * temp;
        tile[r + 8 * hi][t * 16 + col] = 1.f / (1.f + expf(-x));
      }
    }
    __syncthreads();
    if (lane < 16) {
      for (int cc = 0; cc < 64; ++cc) {
        float v = tile[lane][cc];
        int mn = 0; float mv = lv[lane][0];
#pragma unroll
        for (int q = 1; q < KNN; ++q)
          if (lv[lane][q] < mv) { mv = lv[lane][q]; mn = q; }
        if (v > mv) { lv[lane][mn] = v; li[lane][mn] = j0 + cc; }
      }
    }
    __syncthreads();
  }
  if (lane < 16) {
    int row = i0 + lane;
    for (int q = 0; q < KNN; ++q) {
      cand_val[((size_t)row * NPART + part) * KNN + q] = lv[lane][q];
      cand_idx[((size_t)row * NPART + part) * KNN + q] = li[lane][q];
    }
  }
}

// per-row: select global top-20 from 8x20 candidates
__global__ void merge_topk_kernel(const float* cand_val, const int* cand_idx,
                                  float* topv, int* topi) {
  int row = blockIdx.x * blockDim.x + threadIdx.x;
  if (row >= N_PTS) return;
  float v[NPART * KNN]; int ix[NPART * KNN];
  for (int q = 0; q < NPART * KNN; ++q) {
    v[q]  = cand_val[(size_t)row * NPART * KNN + q];
    ix[q] = cand_idx[(size_t)row * NPART * KNN + q];
  }
  for (int s = 0; s < KNN; ++s) {
    int best = 0; float bv = v[0];
    for (int q = 1; q < NPART * KNN; ++q)
      if (v[q] > bv) { bv = v[q]; best = q; }
    topv[(size_t)row * KNN + s] = bv;
    topi[(size_t)row * KNN + s] = ix[best];
    v[best] = -1e30f;
  }
}

// deg[j] = 1 (identity) + #incoming binary edges; also count E = sum(A_bin)
__global__ void deg_scatter_kernel(const float* topv, const int* topi,
                                   float* deg, float* Ecount) {
  int e = blockIdx.x * blockDim.x + threadIdx.x;
  if (e >= N_PTS * KNN) return;
  if (topv[e] > THR) {
    atomicAdd(&deg[topi[e]], 1.f);
    atomicAdd(Ecount, 1.f);
  }
}

__global__ void rsqrt_kernel(const float* deg, float* dinv) {
  int i = blockIdx.x * blockDim.x + threadIdx.x;
  if (i < N_PTS) dinv[i] = rsqrtf(deg[i]);
}

// generic bf16 WMMA GEMM, 4 N-tiles per wave: C[16 x 64] per wave.
// Requires Nc % 64 == 0.
__global__ __launch_bounds__(32) void wmma_gemm4_kernel(
    const unsigned short* A, const unsigned short* B, float* C,
    int M, int Nc, int K) {
  int lane = threadIdx.x;
  int groups_n = Nc / 64;
  int tm = blockIdx.x / groups_n, tg = blockIdx.x % groups_n;
  int i0 = tm * 16, n0 = tg * 64;
  const v8f vzero = {0.f, 0.f, 0.f, 0.f, 0.f, 0.f, 0.f, 0.f};
  v8f acc[4];
#pragma unroll
  for (int t = 0; t < 4; ++t) acc[t] = vzero;
  for (int kt = 0; kt < K; kt += 32) {
    v16bf a = load_frag_row(A, K, i0, kt, lane);
#pragma unroll
    for (int t = 0; t < 4; ++t) {
      v16bf b = load_frag_col(B, Nc, kt, n0 + t * 16, lane);
      acc[t] = wmma_bf16(a, b, acc[t]);
    }
  }
  int hi = lane >> 4, col = lane & 15;
#pragma unroll
  for (int t = 0; t < 4; ++t)
#pragma unroll
    for (int r = 0; r < 8; ++r)
      C[(size_t)(i0 + r + 8 * hi) * Nc + n0 + t * 16 + col] = acc[t][r];
}

// Out = Mn.T @ Y (scatter over sparse rows); Out must be pre-zeroed
__global__ void aggregate_kernel(const float* Y, const int* topi, const float* topv,
                                 const float* dinv, float* Out, int C) {
  int i = blockIdx.x;
  int c = threadIdx.x;
  if (c >= C) return;
  float y  = Y[(size_t)i * C + c];
  float di = dinv[i];
  atomicAdd(&Out[(size_t)i * C + c], di * di * y);  // +I part of M
  for (int e = 0; e < KNN; ++e) {
    float v = topv[(size_t)i * KNN + e];
    if (v > THR) {
      int j = topi[(size_t)i * KNN + e];
      atomicAdd(&Out[(size_t)j * C + c], di * dinv[j] * y);
    }
  }
}

__global__ void bias_relu_kernel(const float* hagg, const float* b1, unsigned short* hb) {
  int i = blockIdx.x, t = threadIdx.x;
  float v = hagg[(size_t)i * HID + t] + b1[t];
  hb[(size_t)i * HID + t] = f2bf(fmaxf(v, 0.f));
}

// softmax over first KMAX cols of sagg(+b2); write S (f32, d_out) and Sb (bf16, padded)
__global__ void softmax_kernel(const float* sagg, const float* b2,
                               float* S, unsigned short* Sb) {
  __shared__ float red[64];
  int i = blockIdx.x, t = threadIdx.x;
  float v = (t < KMAX) ? (sagg[(size_t)i * KPAD + t] + b2[t]) : -1e30f;
  red[t] = v; __syncthreads();
  for (int s = 32; s > 0; s >>= 1) { if (t < s) red[t] = fmaxf(red[t], red[t + s]); __syncthreads(); }
  float mx = red[0]; __syncthreads();
  float e = (t < KMAX) ? expf(v - mx) : 0.f;
  red[t] = e; __syncthreads();
  for (int s = 32; s > 0; s >>= 1) { if (t < s) red[t] += red[t + s]; __syncthreads(); }
  float sv = e / red[0];
  if (t < KMAX) S[(size_t)i * KMAX + t] = sv;
  Sb[(size_t)i * KPAD + t] = f2bf((t < KMAX) ? sv : 0.f);
}

// t = adj @ S (masked top-k values, NOT thresholded), bf16 padded
__global__ void adjS_kernel(const float* S, const float* topv, const int* topi,
                            unsigned short* tb) {
  int i = blockIdx.x, t = threadIdx.x;
  float acc = 0.f;
  if (t < KMAX) {
    for (int e = 0; e < KNN; ++e) {
      float v = topv[(size_t)i * KNN + e];
      int j = topi[(size_t)i * KNN + e];
      acc += v * S[(size_t)j * KMAX + t];
    }
  }
  tb[(size_t)i * KPAD + t] = f2bf((t < KMAX) ? acc : 0.f);
}

// X_t = S.T @ X : A = S.T via column-loader on padded Sb, B = Xb
__global__ __launch_bounds__(32) void xt_gemm_kernel(
    const unsigned short* Sb, const unsigned short* Xb, float* Xt) {
  int lane = threadIdx.x;
  int tiles_n = FEAT / 16;
  int tm = blockIdx.x / tiles_n, tn = blockIdx.x % tiles_n;
  int m0 = tm * 16, n0 = tn * 16;
  v8f c = {0.f, 0.f, 0.f, 0.f, 0.f, 0.f, 0.f, 0.f};
  for (int kt = 0; kt < N_PTS; kt += 32) {
    v16bf a = load_frag_col(Sb, KPAD, kt, m0, lane);  // A[m][k] = S[k][m]
    v16bf b = load_frag_col(Xb, FEAT, kt, n0, lane);
    c = wmma_bf16(a, b, c);
  }
  int hi = lane >> 4, col = lane & 15;
#pragma unroll
  for (int r = 0; r < 8; ++r) {
    int m = m0 + r + 8 * hi;
    if (m < KMAX) Xt[(size_t)m * FEAT + n0 + col] = c[r];
  }
}

// A_t = threshold( S.T @ (adj @ S) )
__global__ __launch_bounds__(32) void acoarse_gemm_kernel(
    const unsigned short* Sb, const unsigned short* tb, float* At) {
  int lane = threadIdx.x;
  int tiles_n = KPAD / 16;
  int tm = blockIdx.x / tiles_n, tn = blockIdx.x % tiles_n;
  int m0 = tm * 16, n0 = tn * 16;
  v8f c = {0.f, 0.f, 0.f, 0.f, 0.f, 0.f, 0.f, 0.f};
  for (int kt = 0; kt < N_PTS; kt += 32) {
    v16bf a = load_frag_col(Sb, KPAD, kt, m0, lane);
    v16bf b = load_frag_col(tb, KPAD, kt, n0, lane);
    c = wmma_bf16(a, b, c);
  }
  int hi = lane >> 4, col = lane & 15;
#pragma unroll
  for (int r = 0; r < 8; ++r) {
    int m = m0 + r + 8 * hi;
    int n = n0 + col;
    if (m < KMAX && n < KMAX) {
      float v = c[r];
      At[(size_t)m * KMAX + n] = (v > THR) ? v : 0.f;
    }
  }
}

// L accumulation: sum over binary edges of d2(i,j) * dot(S_i, S_j)
__global__ void loss_kernel(const float* S, const float* topv, const int* topi,
                            const float* pos, float* Lsum) {
  __shared__ float red[64];
  int i = blockIdx.x, t = threadIdx.x;
  float acc = 0.f;
  float sx = (t < KMAX) ? S[(size_t)i * KMAX + t] : 0.f;
  float pix = pos[2 * i], piy = pos[2 * i + 1];
  for (int e = 0; e < KNN; ++e) {
    float v = topv[(size_t)i * KNN + e];  // uniform across block
    if (v > THR) {
      int j = topi[(size_t)i * KNN + e];
      float dot = (t < KMAX) ? sx * S[(size_t)j * KMAX + t] : 0.f;
      red[t] = dot; __syncthreads();
      for (int s = 32; s > 0; s >>= 1) { if (t < s) red[t] += red[t + s]; __syncthreads(); }
      if (t == 0) {
        float dx = pix - pos[2 * j], dy = piy - pos[2 * j + 1];
        acc += (dx * dx + dy * dy) * red[0];
      }
      __syncthreads();
    }
  }
  if (t == 0) atomicAdd(Lsum, acc);
}

__global__ void finalize_kernel(const float* Lsum, const float* Ecount,
                                const float* spatial_weight, const float* lambda_weight,
                                float* outL, float* outLam) {
  if (threadIdx.x == 0 && blockIdx.x == 0) {
    outL[0]   = spatial_weight[0] * Lsum[0] / fmaxf(Ecount[0], 1.f);
    outLam[0] = 1.f / (1.f + expf(-lambda_weight[0]));
  }
}

// ---------- host launcher ----------
extern "C" void kernel_launch(void* const* d_in, const int* in_sizes, int n_in,
                              void* d_out, int out_size, void* d_ws, size_t ws_size,
                              hipStream_t stream) {
  (void)in_sizes; (void)n_in; (void)out_size; (void)ws_size;
  const float* features       = (const float*)d_in[0];
  const float* positions      = (const float*)d_in[1];
  const float* lambda_weight  = (const float*)d_in[2];
  const float* temperature    = (const float*)d_in[3];
  const float* spatial_weight = (const float*)d_in[4];
  const float* W1 = (const float*)d_in[5];
  const float* b1 = (const float*)d_in[6];
  const float* W2 = (const float*)d_in[7];
  const float* b2 = (const float*)d_in[8];

  char* ws = (char*)d_ws;
  size_t off = 0;
  auto carve = [&](size_t bytes) -> void* {
    void* p = ws + off;
    off = (off + bytes + 255) & ~(size_t)255;
    return p;
  };
  unsigned short* fnb  = (unsigned short*)carve((size_t)N_PTS * FEAT * 2);
  unsigned short* Xb   = (unsigned short*)carve((size_t)N_PTS * FEAT * 2);
  unsigned short* W1b  = (unsigned short*)carve((size_t)FEAT * HID * 2);
  unsigned short* W2b  = (unsigned short*)carve((size_t)HID * KPAD * 2);
  float* cand_val      = (float*)carve((size_t)N_PTS * NPART * KNN * 4);
  int*   cand_idx      = (int*)  carve((size_t)N_PTS * NPART * KNN * 4);
  float* topv          = (float*)carve((size_t)N_PTS * KNN * 4);
  int*   topi          = (int*)  carve((size_t)N_PTS * KNN * 4);
  float* deg           = (float*)carve((size_t)N_PTS * 4);
  float* dinv          = (float*)carve((size_t)N_PTS * 4);
  float* XW            = (float*)carve((size_t)N_PTS * HID * 4);
  float* hagg          = (float*)carve((size_t)N_PTS * HID * 4);
  unsigned short* hb   = (unsigned short*)carve((size_t)N_PTS * HID * 2);
  float* hW2           = (float*)carve((size_t)N_PTS * KPAD * 4);
  float* sagg          = (float*)carve((size_t)N_PTS * KPAD * 4);
  unsigned short* Sb   = (unsigned short*)carve((size_t)N_PTS * KPAD * 2);
  unsigned short* tb   = (unsigned short*)carve((size_t)N_PTS * KPAD * 2);
  float* scal          = (float*)carve(256);
  float* Ecount = scal;
  float* Lsum   = scal + 1;

  float* out    = (float*)d_out;
  float* S_out  = out;                                  // [N, 50]
  float* Xt_out = out + (size_t)N_PTS * KMAX;           // [50, 512]
  float* At_out = Xt_out + (size_t)KMAX * FEAT;         // [50, 50]
  float* L_out  = At_out + (size_t)KMAX * KMAX;         // scalar
  float* lam_out = L_out + 1;                           // scalar

  // 1. weight conversion + feature prep
  convW1_kernel<<<(FEAT * HID + 255) / 256, 256, 0, stream>>>(W1, W1b);
  convW2_kernel<<<(HID * KPAD + 255) / 256, 256, 0, stream>>>(W2, W2b);
  prep_kernel<<<N_PTS, 256, 0, stream>>>(features, fnb, Xb);

  // 2. fused similarity WMMA + streaming top-k (never materializes N x N)
  simtopk_kernel<<<(N_PTS / 16) * NPART, 32, 0, stream>>>(
      fnb, positions, lambda_weight, temperature, cand_val, cand_idx);
  merge_topk_kernel<<<(N_PTS + 255) / 256, 256, 0, stream>>>(cand_val, cand_idx, topv, topi);

  // 3. degrees (M = A_bin + I) and normalization
  fill_kernel<<<(N_PTS + 255) / 256, 256, 0, stream>>>(deg, 1.0f, N_PTS);
  fill_kernel<<<1, 256, 0, stream>>>(scal, 0.0f, 2);
  deg_scatter_kernel<<<(N_PTS * KNN + 255) / 256, 256, 0, stream>>>(topv, topi, deg, Ecount);
  rsqrt_kernel<<<(N_PTS + 255) / 256, 256, 0, stream>>>(deg, dinv);

  // 4. GCN layer 1: XW = X @ W1 (WMMA), aggregate, bias+relu -> hb (bf16)
  wmma_gemm4_kernel<<<(N_PTS / 16) * (HID / 64), 32, 0, stream>>>(Xb, W1b, XW, N_PTS, HID, FEAT);
  fill_kernel<<<(N_PTS * HID + 255) / 256, 256, 0, stream>>>(hagg, 0.0f, N_PTS * HID);
  aggregate_kernel<<<N_PTS, HID, 0, stream>>>(XW, topi, topv, dinv, hagg, HID);
  bias_relu_kernel<<<N_PTS, HID, 0, stream>>>(hagg, b1, hb);

  // 5. GCN layer 2: hW2 = h @ W2 (WMMA, N padded to 64), aggregate, softmax
  wmma_gemm4_kernel<<<(N_PTS / 16) * (KPAD / 64), 32, 0, stream>>>(hb, W2b, hW2, N_PTS, KPAD, HID);
  fill_kernel<<<(N_PTS * KPAD + 255) / 256, 256, 0, stream>>>(sagg, 0.0f, N_PTS * KPAD);
  aggregate_kernel<<<N_PTS, KPAD, 0, stream>>>(hW2, topi, topv, dinv, sagg, KPAD);
  softmax_kernel<<<N_PTS, 64, 0, stream>>>(sagg, b2, S_out, Sb);

  // 6. coarsening: X_t = S.T @ X and A_t = thr(S.T @ (adj @ S)), both via WMMA
  adjS_kernel<<<N_PTS, 64, 0, stream>>>(S_out, topv, topi, tb);
  xt_gemm_kernel<<<(KPAD / 16) * (FEAT / 16), 32, 0, stream>>>(Sb, Xb, Xt_out);
  acoarse_gemm_kernel<<<(KPAD / 16) * (KPAD / 16), 32, 0, stream>>>(Sb, tb, At_out);

  // 7. spatial loss + scalars
  loss_kernel<<<N_PTS, 64, 0, stream>>>(S_out, topv, topi, positions, Lsum);
  finalize_kernel<<<1, 1, 0, stream>>>(Lsum, Ecount, spatial_weight, lambda_weight, L_out, lam_out);
}
